// GPTModel_87290915324641
// MI455X (gfx1250) — matrix operand order repaired and checked by reading, not compiled, not run on hardware
//
#include <hip/hip_runtime.h>

#define Bz 2
#define Tz 1024
#define Ez 1024
#define Hz 16
#define Lz 8
#define Vz 32000
#define CHz 64
#define Mz (Bz * Tz) /* 2048 */

typedef __attribute__((ext_vector_type(16))) __bf16 v16bf;
typedef __attribute__((ext_vector_type(8)))  __bf16 v8bf;
typedef __attribute__((ext_vector_type(8)))  float  v8f;

static __device__ __forceinline__ v16bf cat8(v8bf lo, v8bf hi) {
  return __builtin_shufflevector(lo, hi, 0, 1, 2, 3, 4, 5, 6, 7,
                                 8, 9, 10, 11, 12, 13, 14, 15);
}

static __device__ __forceinline__ v8f wmma_bf16(v16bf a, v16bf b, v8f c) {
  return __builtin_amdgcn_wmma_f32_16x16x32_bf16(false, a, false, b,
                                                 (short)0, c, false, false);
}

// -------------------------------------------------------------------------
// Embedding gather + positional add: h[b,t,:] = emb[x[b,t],:] + pos[t,:]
// -------------------------------------------------------------------------
__global__ __launch_bounds__(256) void embed_k(const int* __restrict__ x,
                                               const float* __restrict__ emb,
                                               const float* __restrict__ pos,
                                               float* __restrict__ h) {
  size_t i = (size_t)blockIdx.x * 256 + threadIdx.x;  // over M*E
  int row = (int)(i >> 10);  // E = 1024
  int e = (int)(i & 1023);
  int t = row & (Tz - 1);
  h[i] = emb[(size_t)x[row] * Ez + e] + pos[(size_t)t * Ez + e];
}

// -------------------------------------------------------------------------
// LayerNorm over last dim (E=1024), one row per 256-thread block.
// -------------------------------------------------------------------------
__global__ __launch_bounds__(256) void layernorm_k(const float* __restrict__ x,
                                                   const float* __restrict__ g,
                                                   const float* __restrict__ b,
                                                   float* __restrict__ y) {
  __shared__ float red[16];
  const int row = blockIdx.x;
  const float* xr = x + (size_t)row * Ez;
  float v[4], s = 0.f, ss = 0.f;
#pragma unroll
  for (int i = 0; i < 4; ++i) {
    v[i] = xr[threadIdx.x + i * 256];
    s += v[i];
    ss += v[i] * v[i];
  }
#pragma unroll
  for (int m = 1; m < 32; m <<= 1) {
    s += __shfl_xor(s, m, 32);
    ss += __shfl_xor(ss, m, 32);
  }
  const int w = threadIdx.x >> 5;
  if ((threadIdx.x & 31) == 0) {
    red[w] = s;
    red[8 + w] = ss;
  }
  __syncthreads();
  float ts = 0.f, tss = 0.f;
#pragma unroll
  for (int i = 0; i < 8; ++i) {
    ts += red[i];
    tss += red[8 + i];
  }
  const float mean = ts * (1.0f / Ez);
  const float var = tss * (1.0f / Ez) - mean * mean;
  const float rstd = rsqrtf(var + 1e-5f);
#pragma unroll
  for (int i = 0; i < 4; ++i) {
    const int e = threadIdx.x + i * 256;
    y[(size_t)row * Ez + e] = (v[i] - mean) * rstd * g[e] + b[e];
  }
}

// -------------------------------------------------------------------------
// Generic GEMM: C[M,N] = act(A[M,K] @ W[K,N] + bias [+ resid]); bf16 WMMA.
// 128x256x32 block tiles, 8 waves (2x4), each wave -> 64x64 (4x4 WMMA
// tiles). Double-buffered LDS; global loads for slab k+1 are issued before
// the WMMA block of slab k; weight slab k+2 is prefetched into L2.
// LDS: A row-major (m x k), B transposed (n x k); row stride 40 bf16 keeps
// every fragment read a 16B-aligned ds_read_b128.
// -------------------------------------------------------------------------
__global__ __launch_bounds__(256) void gemm_bf16_k(
    const float* __restrict__ A, const float* __restrict__ W,
    const float* __restrict__ bias, const float* resid, float* C, int M, int N,
    int K, int relu) {
  __shared__ __align__(16) __bf16 sA[2][128 * 40];
  __shared__ __align__(16) __bf16 sB[2][256 * 40];
  const int tid = threadIdx.x;
  const int w = tid >> 5;
  const int lane = tid & 31;
  const int half = lane >> 4;
  const int lr = lane & 15;
  const int m0 = blockIdx.y * 128;
  const int n0 = blockIdx.x * 256;
  const int wm = (w & 1) * 64;   // wave row base within block tile
  const int wn = (w >> 1) * 64;  // wave col base within block tile

  // Staging coordinates (per thread): A = 4 float4, B = 8 float4.
  const int aR = tid >> 3, aC = (tid & 7) << 2;
  const int bK = tid >> 6, bN = (tid & 63) << 2;
  float4 fa[4], fb[8];

  auto gload = [&](int k0) {
#pragma unroll
    for (int j = 0; j < 4; ++j)
      fa[j] = *(const float4*)(A + (size_t)(m0 + aR + j * 32) * K + k0 + aC);
#pragma unroll
    for (int j = 0; j < 8; ++j)
      fb[j] = *(const float4*)(W + (size_t)(k0 + bK + j * 4) * N + n0 + bN);
  };
  auto lstore = [&](int buf) {
#pragma unroll
    for (int j = 0; j < 4; ++j) {
      __bf16* d = &sA[buf][(aR + j * 32) * 40 + aC];
      d[0] = (__bf16)fa[j].x;
      d[1] = (__bf16)fa[j].y;
      d[2] = (__bf16)fa[j].z;
      d[3] = (__bf16)fa[j].w;
    }
#pragma unroll
    for (int j = 0; j < 8; ++j) {
      const int kk = bK + j * 4;
      __bf16* d = &sB[buf][0];
      d[(bN + 0) * 40 + kk] = (__bf16)fb[j].x;
      d[(bN + 1) * 40 + kk] = (__bf16)fb[j].y;
      d[(bN + 2) * 40 + kk] = (__bf16)fb[j].z;
      d[(bN + 3) * 40 + kk] = (__bf16)fb[j].w;
    }
  };

  v8f acc[4][4];
#pragma unroll
  for (int mt = 0; mt < 4; ++mt)
#pragma unroll
    for (int nt = 0; nt < 4; ++nt) acc[mt][nt] = (v8f){0, 0, 0, 0, 0, 0, 0, 0};

  gload(0);
  lstore(0);
  __syncthreads();

  const int nk = K >> 5;
  for (int kt = 0; kt < nk; ++kt) {
    const int cur = kt & 1;
    if (kt + 1 < nk) {
      gload((kt + 1) << 5);  // overlap global loads with WMMA below
      if (kt + 2 < nk)       // stream next weight slab into L2
        __builtin_prefetch(W + (size_t)(((kt + 2) << 5) + bK) * N + n0 + bN, 0,
                           1);
    }
    v16bf af[4], bf[4];
#pragma unroll
    for (int mt = 0; mt < 4; ++mt) {
      const __bf16* p = &sA[cur][(wm + mt * 16 + lr) * 40];
      af[mt] =
          cat8(*(const v8bf*)(p + half * 8), *(const v8bf*)(p + 16 + half * 8));
    }
#pragma unroll
    for (int nt = 0; nt < 4; ++nt) {
      const __bf16* p = &sB[cur][(wn + nt * 16 + lr) * 40];
      bf[nt] =
          cat8(*(const v8bf*)(p + half * 8), *(const v8bf*)(p + 16 + half * 8));
    }
#pragma unroll
    for (int mt = 0; mt < 4; ++mt)
#pragma unroll
      for (int nt = 0; nt < 4; ++nt)
        acc[mt][nt] = wmma_bf16(af[mt], bf[nt], acc[mt][nt]);
    if (kt + 1 < nk) lstore(cur ^ 1);
    __syncthreads();
  }

  // Epilogue: bias hoisted per N-tile; residual is a uniform branch.
  float bv[4];
#pragma unroll
  for (int nt = 0; nt < 4; ++nt)
    bv[nt] = bias ? bias[n0 + wn + nt * 16 + lr] : 0.f;

  if (resid) {
#pragma unroll
    for (int mt = 0; mt < 4; ++mt)
#pragma unroll
      for (int r = 0; r < 8; ++r) {
        const size_t grow = m0 + wm + mt * 16 + r + 8 * half;
#pragma unroll
        for (int nt = 0; nt < 4; ++nt) {
          const int gcol = n0 + wn + nt * 16 + lr;
          C[grow * N + gcol] = acc[mt][nt][r] + bv[nt] + resid[grow * N + gcol];
        }
      }
  } else {
#pragma unroll
    for (int mt = 0; mt < 4; ++mt)
#pragma unroll
      for (int r = 0; r < 8; ++r) {
        const size_t grow = m0 + wm + mt * 16 + r + 8 * half;
#pragma unroll
        for (int nt = 0; nt < 4; ++nt) {
          const int gcol = n0 + wn + nt * 16 + lr;
          float v = acc[mt][nt][r] + bv[nt];
          if (relu) v = fmaxf(v, 0.f);
          C[grow * N + gcol] = v;
        }
      }
  }
}

// -------------------------------------------------------------------------
// Flash attention: one wave per (b, h, 16-row Q tile). kqv layout per
// reference: (B, T, H, Ch, 3) with k/q/v at last index 0/1/2.
// Output layout: out[b, t, c*H + h]  (channel-major over heads).
// -------------------------------------------------------------------------
__global__ __launch_bounds__(128) void attn_k(const float* __restrict__ kqv,
                                              float* __restrict__ out) {
  __shared__ __align__(16) __bf16 pst[4][16 * 40];
  const int w = threadIdx.x >> 5;
  const int lane = threadIdx.x & 31;
  const int half = lane >> 4;
  const int lr = lane & 15;
  const int gw = blockIdx.x * 4 + w;
  const int t0 = (gw & 63) << 4;  // 64 row-tiles per (b,h)
  const int bh = gw >> 6;
  const int h = bh & (Hz - 1);
  const int b = bh >> 4;
  const float* base = kqv + (size_t)b * Tz * (3 * Ez) + h * (CHz * 3);

  // Q tile as two A fragments (K = c0..31 and c32..63), resident in VGPRs.
  v16bf qf[2];
#pragma unroll
  for (int kb = 0; kb < 2; ++kb) {
    const float* qrow = base + (size_t)(t0 + lr) * (3 * Ez);
    v16bf a;
#pragma unroll
    for (int i = 0; i < 16; ++i) {
      const int c =
          kb * 32 + ((i < 8) ? (half * 8 + i) : (16 + half * 8 + (i - 8)));
      a[i] = (__bf16)qrow[c * 3 + 1];  // q at slice index 1
    }
    qf[kb] = a;
  }

  v8f o[4];
#pragma unroll
  for (int nt = 0; nt < 4; ++nt) o[nt] = (v8f){0, 0, 0, 0, 0, 0, 0, 0};
  float mrun[8], lrun[8];
#pragma unroll
  for (int r = 0; r < 8; ++r) {
    mrun[r] = -__builtin_inff();
    lrun[r] = 0.f;
  }
  const float scale = 0.125f;  // 64^-0.5

  for (int s0 = 0; s0 < t0 + 16; s0 += 32) {
    float sv[2][8];
#pragma unroll
    for (int ti = 0; ti < 2; ++ti) {
      v8f s8 = (v8f){0, 0, 0, 0, 0, 0, 0, 0};
#pragma unroll
      for (int kb = 0; kb < 2; ++kb) {
        const float* krow = base + (size_t)(s0 + ti * 16 + lr) * (3 * Ez);
        v16bf bv;
#pragma unroll
        for (int i = 0; i < 16; ++i) {
          const int c =
              kb * 32 + ((i < 8) ? (half * 8 + i) : (16 + half * 8 + (i - 8)));
          bv[i] = (__bf16)krow[c * 3 + 0];  // k at slice index 0
        }
        s8 = wmma_bf16(qf[kb], bv, s8);
      }
#pragma unroll
      for (int r = 0; r < 8; ++r) {
        const int gr = t0 + r + 8 * half;
        const int gc = s0 + ti * 16 + lr;
        sv[ti][r] = (gc > gr) ? -__builtin_inff() : s8[r] * scale;
      }
    }
    // Online softmax: rows live across the 16 lanes of each half.
#pragma unroll
    for (int r = 0; r < 8; ++r) {
      float mx = fmaxf(sv[0][r], sv[1][r]);
#pragma unroll
      for (int d = 1; d < 16; d <<= 1) mx = fmaxf(mx, __shfl_xor(mx, d, 32));
      const float mnew = fmaxf(mrun[r], mx);
      const float alpha = __expf(mrun[r] - mnew);
      const float p0 = __expf(sv[0][r] - mnew);
      const float p1 = __expf(sv[1][r] - mnew);
      float ls = p0 + p1;
#pragma unroll
      for (int d = 1; d < 16; d <<= 1) ls += __shfl_xor(ls, d, 32);
      lrun[r] = lrun[r] * alpha + ls;
      mrun[r] = mnew;
      sv[0][r] = p0;
      sv[1][r] = p1;
#pragma unroll
      for (int nt = 0; nt < 4; ++nt) o[nt][r] *= alpha;
    }
    // C-layout -> A-layout for P via per-wave LDS staging.
    __bf16* pl = &pst[w][0];
#pragma unroll
    for (int ti = 0; ti < 2; ++ti)
#pragma unroll
      for (int r = 0; r < 8; ++r)
        pl[(r + 8 * half) * 40 + ti * 16 + lr] = (__bf16)sv[ti][r];
    asm volatile("s_wait_dscnt 0x0" ::: "memory");
    v16bf pa;
    {
      const __bf16* p = pl + lr * 40;
      const v8bf lo = *(const v8bf*)(p + half * 8);
      const v8bf hi = *(const v8bf*)(p + 16 + half * 8);
      pa = cat8(lo, hi);
    }
    asm volatile("" ::: "memory");
    // O(16x64) += P(16x32) @ V(32x64)
#pragma unroll
    for (int nt = 0; nt < 4; ++nt) {
      v16bf bv;
#pragma unroll
      for (int i = 0; i < 16; ++i) {
        const int srow = s0 + ((i < 8) ? (half * 8 + i) : (16 + half * 8 + (i - 8)));
        const float* vrow = base + (size_t)srow * (3 * Ez);
        bv[i] = (__bf16)vrow[(nt * 16 + lr) * 3 + 2];  // v at slice index 2
      }
      o[nt] = wmma_bf16(pa, bv, o[nt]);
    }
  }

#pragma unroll
  for (int r = 0; r < 8; ++r) {
    const float inv = 1.0f / lrun[r];
    const int t = t0 + r + 8 * half;
#pragma unroll
    for (int nt = 0; nt < 4; ++nt) {
      const int c = nt * 16 + lr;
      out[((size_t)(b * Tz) + t) * Ez + c * Hz + h] = o[nt][r] * inv;
    }
  }
}

// -------------------------------------------------------------------------
extern "C" void kernel_launch(void* const* d_in, const int* in_sizes, int n_in,
                              void* d_out, int out_size, void* d_ws,
                              size_t ws_size, hipStream_t stream) {
  const int* x = (const int*)d_in[0];
  const float* emb = (const float*)d_in[1];
  const float* pos = (const float*)d_in[2];
  const float* ln1_g = (const float*)d_in[3];
  const float* ln1_b = (const float*)d_in[4];
  const float* Wu = (const float*)d_in[5];
  const float* Wp = (const float*)d_in[6];
  const float* bp = (const float*)d_in[7];
  const float* ln2_g = (const float*)d_in[8];
  const float* ln2_b = (const float*)d_in[9];
  const float* W1 = (const float*)d_in[10];
  const float* b1 = (const float*)d_in[11];
  const float* W2 = (const float*)d_in[12];
  const float* b2 = (const float*)d_in[13];
  const float* lnf_g = (const float*)d_in[14];
  const float* lnf_b = (const float*)d_in[15];
  const float* Wout = (const float*)d_in[16];
  const float* bout = (const float*)d_in[17];
  float* out = (float*)d_out;

  float* h = (float*)d_ws;                 // M*E
  float* xn = h + (size_t)Mz * Ez;         // M*E
  float* kqv = xn + (size_t)Mz * Ez;       // M*3E
  float* att = kqv + (size_t)Mz * 3 * Ez;  // M*E
  float* mlp = att + (size_t)Mz * Ez;      // M*4E

  embed_k<<<(Mz * Ez) / 256, 256, 0, stream>>>(x, emb, pos, h);

  for (int l = 0; l < Lz; ++l) {
    layernorm_k<<<Mz, 256, 0, stream>>>(h, ln1_g + (size_t)l * Ez,
                                        ln1_b + (size_t)l * Ez, xn);
    gemm_bf16_k<<<dim3(3 * Ez / 256, Mz / 128), 256, 0, stream>>>(
        xn, Wu + (size_t)l * Ez * 3 * Ez, nullptr, nullptr, kqv, Mz, 3 * Ez,
        Ez, 0);
    attn_k<<<(Bz * Hz * (Tz / 16)) / 4, 128, 0, stream>>>(kqv, att);
    gemm_bf16_k<<<dim3(Ez / 256, Mz / 128), 256, 0, stream>>>(
        att, Wp + (size_t)l * Ez * Ez, bp + (size_t)l * Ez, h, h, Mz, Ez, Ez,
        0);
    layernorm_k<<<Mz, 256, 0, stream>>>(h, ln2_g + (size_t)l * Ez,
                                        ln2_b + (size_t)l * Ez, xn);
    gemm_bf16_k<<<dim3(4 * Ez / 256, Mz / 128), 256, 0, stream>>>(
        xn, W1 + (size_t)l * Ez * 4 * Ez, b1 + (size_t)l * 4 * Ez, nullptr,
        mlp, Mz, 4 * Ez, Ez, 1);
    gemm_bf16_k<<<dim3(Ez / 256, Mz / 128), 256, 0, stream>>>(
        mlp, W2 + (size_t)l * 4 * Ez * Ez, b2 + (size_t)l * Ez, h, h, Mz, Ez,
        4 * Ez, 0);
  }

  layernorm_k<<<Mz, 256, 0, stream>>>(h, lnf_g, lnf_b, xn);
  gemm_bf16_k<<<dim3(Vz / 256, Mz / 128), 256, 0, stream>>>(
      xn, Wout, bout, nullptr, out, Mz, Vz, Ez, 0);
}
